// HierarchicalVQVAE_43894565765542
// MI455X (gfx1250) — matrix operand order repaired and checked by reading, not compiled, not run on hardware
//
#include <hip/hip_runtime.h>
#include <hip/hip_bf16.h>
#include <math.h>

typedef __bf16 bf16;
typedef bf16  bf16x8  __attribute__((ext_vector_type(8)));
typedef bf16  bf16x16 __attribute__((ext_vector_type(16)));
typedef float f32x8   __attribute__((ext_vector_type(8)));

__device__ __forceinline__ bf16 f2bf(float f) {
  union { float f; unsigned u; } v; v.f = f;
  unsigned r = v.u + 0x7FFFu + ((v.u >> 16) & 1u);   // round-to-nearest-even
  unsigned short h = (unsigned short)(r >> 16);
  return __builtin_bit_cast(bf16, h);
}
__device__ __forceinline__ float bf2f(bf16 b) {
  unsigned short h = __builtin_bit_cast(unsigned short, b);
  union { unsigned u; float f; } v; v.u = ((unsigned)h) << 16;
  return v.f;
}
__device__ __forceinline__ bf16x16 join8(bf16x8 a, bf16x8 b) {
  return __builtin_shufflevector(a, b, 0, 1, 2, 3, 4, 5, 6, 7,
                                       8, 9, 10, 11, 12, 13, 14, 15);
}

// ---------------------------------------------------------------------------
// Weight repack -> bf16 B matrix, COLUMN-major [Npad][Kpad] (K contiguous).
//   k = cin*16 + ky*4 + kx, n = cout. Zero outside [K, Cout].
//   isT=0: w is OIHW (conv). isT=1: w is IOHW (convT gather form).
// ---------------------------------------------------------------------------
__global__ void repack_w(const float* __restrict__ w, bf16* __restrict__ outB,
                         int Cout, int Cin, int Kpad, int Npad, int isT)
{
  int i = blockIdx.x * 256 + threadIdx.x;
  int total = Kpad * Npad;
  if (i >= total) return;
  int n = i / Kpad, k = i - (i / Kpad) * Kpad;
  float v = 0.f;
  int K = Cin * 16;
  if (k < K && n < Cout) {
    int cin = k >> 4, kk = k & 15;
    v = isT ? w[((size_t)cin * Cout + n) * 16 + kk]
            : w[((size_t)n * Cin + cin) * 16 + kk];
  }
  outB[(size_t)n * Kpad + k] = f2bf(v);
}

__global__ void init_loss(float* p) { if (threadIdx.x == 0) *p = 0.f; }

// ---------------------------------------------------------------------------
// Implicit-GEMM conv / convT with bf16 WMMA, fp32 accumulate.
//   mode 0: conv k=4, pad=1, stride s   : iy = oy*s - 1 + ky
//   mode 1: convT k=4, stride=2, pad=1  : iy = (oy+1-ky)/2 when parity ok
// A: [M = B*OH*OW, K = Cin*16] gathered from NCHW fp32 -> bf16 LDS (b128 stores)
//    Gather pattern (tap offsets + validity mask) hoisted out of the K loop:
//    steady state is branchless (clamped loads + select-to-zero).
// B: prepacked bf16 column-major [Npad][Kpad]; tile fetched with
//    global_load_async_to_lds_b128 (ASYNCcnt) straight into LDS.
// Block: 256 thr = 8 waves; tile 64x64; wave = 16 rows x 32 cols (2 wmma/step).
// ---------------------------------------------------------------------------
#define BM 64
#define BN 64
#define BK 32

__global__ void __launch_bounds__(256)
conv_gemm_wmma(const float* __restrict__ in, const bf16* __restrict__ BmatT,
               const float* __restrict__ bias, float* __restrict__ out,
               int Bn, int Cin, int H, int W, int OH, int OW,
               int Nreal, int Npad, int K, int Kpad,
               int stride, int mode, int act, int chanOff, int Cstride)
{
  __shared__ bf16 As [BM][BK + 8];   // 80B row stride: 16B aligned, bank-skewed
  __shared__ bf16 BsT[BN][BK + 8];   // [col][k], same stride

  const int tid  = threadIdx.x;
  const int lane = tid & 31;
  const int wave = tid >> 5;
  const int wm   = wave & 3;
  const int wn   = wave >> 2;
  const int hi   = lane >> 4;
  const int l15  = lane & 15;

  const int OHW = OH * OW;
  const int M   = Bn * OHW;
  const int bm0 = blockIdx.y * BM;
  const int bn0 = blockIdx.x * BN;

  // per-thread tile coords: one row / one column, one 8-element K segment
  const int mmA = tid >> 2;            // 0..63  (A row)
  const int ksA = (tid & 3) << 3;      // 0,8,16,24
  const int nnB = tid >> 2;            // 0..63  (B column)

  // ---- hoisted per-thread im2col pattern (loop-invariant) ----
  // decode this thread's A row once
  const int mA  = bm0 + mmA;
  const bool mOk = mA < M;
  const int mC  = mOk ? mA : (M - 1);
  const int bA  = mC / OHW;
  const int remA = mC - bA * OHW;
  const int oyA = remA / OW;
  const int oxA = remA - oyA * OW;

  unsigned vmask = 0;          // per-tap validity bits
  int poff[8];                 // clamped iy*W+ix per tap
  {
    const int t0 = ksA & 15;   // (k0 is a multiple of 32 -> pattern is K-invariant)
    #pragma unroll
    for (int j = 0; j < 8; ++j) {
      int t  = t0 + j;
      int ky = t >> 2, kx = t & 3;
      int iy, ix;
      bool ok = mOk;
      if (mode == 0) {
        iy = oyA * stride - 1 + ky;
        ix = oxA * stride - 1 + kx;
      } else {
        int ty = oyA + 1 - ky, tx = oxA + 1 - kx;
        ok = ok && (((ty | tx) & 1) == 0);
        iy = ty >> 1; ix = tx >> 1;
      }
      ok = ok && ((unsigned)iy < (unsigned)H) && ((unsigned)ix < (unsigned)W);
      iy = min(max(iy, 0), H - 1);
      ix = min(max(ix, 0), W - 1);
      poff[j] = iy * W + ix;
      if (ok) vmask |= (1u << j);
    }
  }
  const int cinStep = ksA >> 4;        // cin = 2*kt + cinStep
  const size_t planeHW = (size_t)H * W;

  f32x8 acc0 = {};
  f32x8 acc1 = {};

  const int nk = Kpad / BK;
  for (int kt = 0; kt < nk; ++kt) {
    const int k0 = kt * BK;

    // ---- B tile: async global -> LDS, 16B per lane (CDNA5 ASYNC path) ----
    {
      const bf16* gp = BmatT + (size_t)(bn0 + nnB) * Kpad + k0 + ksA;
      unsigned long long ga = (unsigned long long)(uintptr_t)gp;
      unsigned lp = (unsigned)(uintptr_t)(&BsT[nnB][ksA]);
      asm volatile("global_load_async_to_lds_b128 %0, %1, off"
                   :: "v"(lp), "v"(ga) : "memory");
    }

    // ---- A tile: branchless gather, pack 8 bf16, one ds_store_b128 ----
    {
      int cin = (kt << 1) + cinStep;
      unsigned em = ((k0 + ksA) < K) ? vmask : 0u;     // fold K-padding
      int cinC = min(cin, Cin - 1);
      const float* basep = in + ((size_t)bA * Cin + cinC) * planeHW;
      bf16x8 pk;
      #pragma unroll
      for (int j = 0; j < 8; ++j) {
        float raw = basep[poff[j]];                    // clamped, always safe
        float v = ((em >> j) & 1u) ? raw : 0.f;
        pk[j] = f2bf(v);
      }
      *(bf16x8*)(&As[mmA][ksA]) = pk;
    }

    if (k0 + BK < Kpad)
      __builtin_prefetch(BmatT + (size_t)(bn0 + nnB) * Kpad + k0 + BK, 0, 1);

    asm volatile("s_wait_asynccnt 0x0" ::: "memory");
    __syncthreads();

    // ---- fragments: pure ds_load_b128 + register shuffles ----
    const int rowA = wm * 16 + l15;
    const int colB = wn * 32 + l15;
    bf16x16 af  = join8(*(const bf16x8*)(&As[rowA][hi * 8]),
                        *(const bf16x8*)(&As[rowA][16 + hi * 8]));
    bf16x16 bf0 = join8(*(const bf16x8*)(&BsT[colB][hi * 8]),
                        *(const bf16x8*)(&BsT[colB][16 + hi * 8]));
    bf16x16 bf1 = join8(*(const bf16x8*)(&BsT[colB + 16][hi * 8]),
                        *(const bf16x8*)(&BsT[colB + 16][16 + hi * 8]));
    acc0 = __builtin_amdgcn_wmma_f32_16x16x32_bf16(false, af, false, bf0, (short)0, acc0, false, false);
    acc1 = __builtin_amdgcn_wmma_f32_16x16x32_bf16(false, af, false, bf1, (short)0, acc1, false, false);
    __syncthreads();
  }

  // ---- fused epilogue: bias + activation + NCHW scatter (with concat) ----
  #pragma unroll
  for (int r = 0; r < 8; ++r) {
    int row = r + hi * 8;
    int m   = bm0 + wm * 16 + row;
    if (m >= M) continue;
    int b   = m / OHW;
    int rem = m - b * OHW;
    #pragma unroll
    for (int t = 0; t < 2; ++t) {
      int n = bn0 + wn * 32 + t * 16 + l15;
      if (n >= Nreal) continue;
      float v = (t ? acc1[r] : acc0[r]) + bias[n];
      if (act == 1)      v = fmaxf(v, 0.f);
      else if (act == 2) v = 1.f / (1.f + __expf(-v));
      out[((size_t)b * Cstride + chanOff + n) * OHW + rem] = v;
    }
  }
}

// ---------------------------------------------------------------------------
// VQ nearest-code search: dist = |z|^2 + |e|^2 - 2 z.e  (z.e via WMMA)
// 256 thr = 8 waves; each wave owns 16 points; codebook tiled 64-at-a-time.
// LDS rows padded to 136 bf16 (272B: 16B aligned, 4-bank skew).
// ---------------------------------------------------------------------------
__global__ void __launch_bounds__(256)
vq_argmin_wmma(const float* __restrict__ z, const float* __restrict__ emb,
               int* __restrict__ idxOut, int Npts, int HW, int Kcodes)
{
  __shared__ bf16  embS[64][136];
  __shared__ float enS[64];
  __shared__ bf16  zS[8][16][136];
  __shared__ float znS[8][16];

  const int tid   = threadIdx.x;
  const int lane  = tid & 31;
  const int wave  = tid >> 5;
  const int hi    = lane >> 4;
  const int l15   = lane & 15;
  const int mBase = blockIdx.x * 128 + wave * 16;

  // stage this wave's 16 z-vectors (channel-gathered from NCHW), b128 stores
  for (int s = lane; s < 256; s += 32) {
    int p  = s >> 4;
    int cs = (s & 15) << 3;
    int m  = mBase + p;
    bf16x8 pk;
    if (m < Npts) {
      int b = m / HW; int rem = m - b * HW;
      const float* zp = z + ((size_t)b * 128 + cs) * HW + rem;
      #pragma unroll
      for (int j = 0; j < 8; ++j) pk[j] = f2bf(zp[(size_t)j * HW]);
    } else {
      #pragma unroll
      for (int j = 0; j < 8; ++j) pk[j] = f2bf(0.f);
    }
    *(bf16x8*)(&zS[wave][p][cs]) = pk;
  }
  __syncthreads();
  if (lane < 16) {
    float s = 0.f;
    for (int c = 0; c < 128; ++c) { float t = bf2f(zS[wave][lane][c]); s += t * t; }
    znS[wave][lane] = s;
  }
  __syncthreads();

  float minv[8]; int mini[8];
  #pragma unroll
  for (int r = 0; r < 8; ++r) { minv[r] = 3.4e38f; mini[r] = 0; }

  for (int ct = 0; ct < Kcodes / 64; ++ct) {
    // stage 64 codes (fp32 -> bf16), vectorized LDS stores
    for (int s = tid; s < 64 * 16; s += 256) {
      int cc = s >> 4;
      int cs = (s & 15) << 3;
      const float* ep = emb + (size_t)(ct * 64 + cc) * 128 + cs;
      bf16x8 pk;
      #pragma unroll
      for (int j = 0; j < 8; ++j) pk[j] = f2bf(ep[j]);
      *(bf16x8*)(&embS[cc][cs]) = pk;
    }
    __syncthreads();
    if (tid < 64) {
      float s = 0.f;
      const float* e = emb + (size_t)(ct * 64 + tid) * 128;
      for (int c = 0; c < 128; ++c) s += e[c] * e[c];
      enS[tid] = s;
    }
    __syncthreads();

    #pragma unroll
    for (int sub = 0; sub < 4; ++sub) {
      f32x8 acc = {};
      #pragma unroll
      for (int kc = 0; kc < 4; ++kc) {
        bf16x16 af = join8(*(const bf16x8*)(&zS[wave][l15][kc * 32 + hi * 8]),
                           *(const bf16x8*)(&zS[wave][l15][kc * 32 + 16 + hi * 8]));
        bf16x16 bfb = join8(*(const bf16x8*)(&embS[sub * 16 + l15][kc * 32 + hi * 8]),
                            *(const bf16x8*)(&embS[sub * 16 + l15][kc * 32 + 16 + hi * 8]));
        acc = __builtin_amdgcn_wmma_f32_16x16x32_bf16(false, af, false, bfb, (short)0, acc, false, false);
      }
      int code = ct * 64 + sub * 16 + l15;
      float en = enS[sub * 16 + l15];
      #pragma unroll
      for (int r = 0; r < 8; ++r) {
        int row = r + hi * 8;
        float d = znS[wave][row] + en - 2.f * acc[r];
        if (d < minv[r]) { minv[r] = d; mini[r] = code; }
      }
    }
    __syncthreads();
  }

  // 16-lane argmin butterfly (wave32, width-16 halves match C-matrix layout)
  #pragma unroll
  for (int off = 8; off > 0; off >>= 1) {
    #pragma unroll
    for (int r = 0; r < 8; ++r) {
      float ov = __shfl_xor(minv[r], off, 16);
      int   oi = __shfl_xor(mini[r], off, 16);
      if (ov < minv[r]) { minv[r] = ov; mini[r] = oi; }
    }
  }
  if (l15 == 0) {
    #pragma unroll
    for (int r = 0; r < 8; ++r) {
      int m = mBase + r + hi * 8;
      if (m < Npts) idxOut[m] = mini[r];
    }
  }
}

// ---------------------------------------------------------------------------
// VQ gather (straight-through value == zq) + 1.5*MSE loss accumulation.
// ---------------------------------------------------------------------------
__global__ void __launch_bounds__(256)
vq_gather_loss(const float* __restrict__ z, const float* __restrict__ emb,
               const int* __restrict__ idx, float* __restrict__ out,
               int chanOff, int Cstride, int HW, int Npts,
               float scale, float* __restrict__ lossAcc)
{
  int m = blockIdx.x * 256 + threadIdx.x;
  float s = 0.f;
  if (m < Npts) {
    int b = m / HW; int rem = m - b * HW;
    int id = idx[m];
    for (int c = 0; c < 128; ++c) {
      float q  = emb[(size_t)id * 128 + c];
      float zv = z[((size_t)b * 128 + c) * HW + rem];
      float dd = q - zv; s += dd * dd;
      out[((size_t)b * Cstride + chanOff + c) * HW + rem] = q;
    }
  }
  __shared__ float red[256];
  red[threadIdx.x] = s * scale;
  __syncthreads();
  for (int st = 128; st > 0; st >>= 1) {
    if (threadIdx.x < st) red[threadIdx.x] += red[threadIdx.x + st];
    __syncthreads();
  }
  if (threadIdx.x == 0) atomicAdd(lossAcc, red[0]);
}

// ---------------------------------------------------------------------------
extern "C" void kernel_launch(void* const* d_in, const int* in_sizes, int n_in,
                              void* d_out, int out_size, void* d_ws, size_t ws_size,
                              hipStream_t stream)
{
  (void)in_sizes; (void)n_in; (void)ws_size;
  const float* x       = (const float*)d_in[0];
  const float* we_b1   = (const float*)d_in[1];
  const float* be_b1   = (const float*)d_in[2];
  const float* we_b2   = (const float*)d_in[3];
  const float* be_b2   = (const float*)d_in[4];
  const float* we_t    = (const float*)d_in[5];
  const float* be_t    = (const float*)d_in[6];
  const float* emb_top = (const float*)d_in[7];
  const float* emb_bot = (const float*)d_in[8];
  const float* wd_t    = (const float*)d_in[9];
  const float* bd_t    = (const float*)d_in[10];
  const float* wd1     = (const float*)d_in[11];
  const float* bd1     = (const float*)d_in[12];
  const float* wd2     = (const float*)d_in[13];
  const float* bd2     = (const float*)d_in[14];

  size_t off = 0;
  auto alloc = [&](size_t bytes) -> void* {
    void* p = (char*)d_ws + off;
    off += (bytes + 255) & ~(size_t)255;
    return p;
  };
  bf16* wB1  = (bf16*)alloc(64ull   * 64  * 2);
  bf16* wB2  = (bf16*)alloc(1024ull * 128 * 2);
  bf16* wB3  = (bf16*)alloc(2048ull * 128 * 2);
  bf16* wBt  = (bf16*)alloc(2048ull * 128 * 2);
  bf16* wBd1 = (bf16*)alloc(4096ull * 64  * 2);
  bf16* wBd2 = (bf16*)alloc(1024ull * 64  * 2);
  float* a1   = (float*)alloc(16ull * 64  * 128 * 128 * 4);
  float* zb   = (float*)alloc(16ull * 128 * 64  * 64  * 4);
  float* zt   = (float*)alloc(16ull * 128 * 32  * 32  * 4);
  float* ztq  = (float*)alloc(16ull * 128 * 32  * 32  * 4);
  float* comb = (float*)alloc(16ull * 256 * 64  * 64  * 4);
  float* h    = (float*)alloc(16ull * 64  * 128 * 128 * 4);
  int* idxT   = (int*)alloc(16384ull * 4);
  int* idxB   = (int*)alloc(65536ull * 4);

  float* lossPtr = (float*)d_out + (out_size - 1);

  auto rp = [&](const float* w, bf16* outB, int Cout, int Cin, int Kpad, int Npad, int isT) {
    int total = Kpad * Npad;
    repack_w<<<(total + 255) / 256, 256, 0, stream>>>(w, outB, Cout, Cin, Kpad, Npad, isT);
  };
  rp(we_b1, wB1, 64, 3, 64, 64, 0);
  rp(we_b2, wB2, 128, 64, 1024, 128, 0);
  rp(we_t,  wB3, 128, 128, 2048, 128, 0);
  rp(wd_t,  wBt, 128, 128, 2048, 128, 1);
  rp(wd1,   wBd1, 64, 256, 4096, 64, 1);
  rp(wd2,   wBd2, 3, 64, 1024, 64, 1);

  init_loss<<<1, 32, 0, stream>>>(lossPtr);

  auto gemm = [&](const float* in, const bf16* Bm, const float* bias, float* out,
                  int Bn, int Cin, int H, int W, int OH, int OW,
                  int Nreal, int Npad, int K, int Kpad, int stride, int mode, int act,
                  int chanOff, int Cstride) {
    int M = Bn * OH * OW;
    dim3 g(Npad / 64, (M + 63) / 64);
    conv_gemm_wmma<<<g, 256, 0, stream>>>(in, Bm, bias, out, Bn, Cin, H, W, OH, OW,
                                          Nreal, Npad, K, Kpad, stride, mode, act,
                                          chanOff, Cstride);
  };

  // encoder_bottom
  gemm(x,  wB1, be_b1, a1, 16, 3,   256, 256, 128, 128, 64,  64,  48,   64,   2, 0, 1, 0, 64);
  gemm(a1, wB2, be_b2, zb, 16, 64,  128, 128, 64,  64,  128, 128, 1024, 1024, 2, 0, 1, 0, 128);
  // encoder_top
  gemm(zb, wB3, be_t,  zt, 16, 128, 64,  64,  32,  32,  128, 128, 2048, 2048, 2, 0, 1, 0, 128);

  // VQ top
  vq_argmin_wmma<<<(16384 + 127) / 128, 256, 0, stream>>>(zt, emb_top, idxT, 16384, 32 * 32, 512);
  vq_gather_loss<<<(16384 + 255) / 256, 256, 0, stream>>>(zt, emb_top, idxT, ztq, 0, 128,
                                                          32 * 32, 16384,
                                                          1.5f / (16384.f * 128.f), lossPtr);
  // decoder_top: convT(128->128) -> combined channels [128,256)
  gemm(ztq, wBt, bd_t, comb, 16, 128, 32, 32, 64, 64, 128, 128, 2048, 2048, 2, 1, 1, 128, 256);

  // VQ bottom -> combined channels [0,128)
  vq_argmin_wmma<<<(65536 + 127) / 128, 256, 0, stream>>>(zb, emb_bot, idxB, 65536, 64 * 64, 512);
  vq_gather_loss<<<(65536 + 255) / 256, 256, 0, stream>>>(zb, emb_bot, idxB, comb, 0, 256,
                                                          64 * 64, 65536,
                                                          1.5f / (65536.f * 128.f), lossPtr);

  // decoder
  gemm(comb, wBd1, bd1, h, 16, 256, 64, 64, 128, 128, 64, 64, 4096, 4096, 2, 1, 1, 0, 64);
  gemm(h, wBd2, bd2, (float*)d_out, 16, 64, 128, 128, 256, 256, 3, 64, 1024, 1024, 2, 1, 2, 0, 3);
}
